// GroupedQueryAttention_42125039239361
// MI455X (gfx1250) — matrix operand order repaired and checked by reading, not compiled, not run on hardware
//
#include <hip/hip_runtime.h>

// ---------------------------------------------------------------------------
// GQA attention (RoPE + flash attention) for MI455X / gfx1250, wave32 + WMMA.
// Matrix math: v_wmma_f32_16x16x32_bf16 (f32 accumulate).
// Data movement: global_load_async_to_lds_b128 double-buffered LDS staging,
// synchronized with s_wait_asynccnt + workgroup barriers.
// ---------------------------------------------------------------------------

typedef __bf16 bf16_t;
typedef __attribute__((ext_vector_type(8)))  __bf16 v8bf;
typedef __attribute__((ext_vector_type(16))) __bf16 v16bf;
typedef __attribute__((ext_vector_type(8)))  float  v8f;

#define BATCH    4
#define SEQ      2048
#define N_EMBD   2048
#define N_HEAD   16
#define N_KV     4
#define HEAD_DIM 128
#define NREP     (N_HEAD / N_KV)
#define KV_C     (N_KV * HEAD_DIM)   // 512

__device__ __forceinline__ v16bf join16(v8bf lo, v8bf hi) {
  union { v16bf v; v8bf h[2]; } u;
  u.h[0] = lo; u.h[1] = hi;
  return u.v;
}

__device__ __forceinline__ v8f v8f_zero() {
  v8f z;
#pragma unroll
  for (int i = 0; i < 8; ++i) z[i] = 0.0f;
  return z;
}

// 16-byte async DMA global -> LDS (CDNA5). LDS operand is the low 32 bits of
// the generic pointer (== addrspace(3) byte offset).
__device__ __forceinline__ void async_copy16(bf16_t* lds_dst, const bf16_t* gsrc) {
  asm volatile("global_load_async_to_lds_b128 %0, %1, off"
               :: "v"((unsigned int)(unsigned long long)lds_dst),
                  "v"((unsigned long long)gsrc)
               : "memory");
}
__device__ __forceinline__ void wait_async_le4() {
  asm volatile("s_wait_asynccnt 0x4" ::: "memory");
}
__device__ __forceinline__ void wait_async_le1() {
  asm volatile("s_wait_asynccnt 0x1" ::: "memory");
}
__device__ __forceinline__ void wait_async_0() {
  asm volatile("s_wait_asynccnt 0x0" ::: "memory");
}

// ---------------------------------------------------------------------------
// f32 -> bf16 cast
// ---------------------------------------------------------------------------
__global__ void cast_f32_to_bf16(const float* __restrict__ in,
                                 bf16_t* __restrict__ out, int n) {
  int i = blockIdx.x * blockDim.x + threadIdx.x;
  if (i < n) out[i] = (bf16_t)in[i];
}

// ---------------------------------------------------------------------------
// C[M,N] (f32) = A[M,K] (bf16, row-major) @ W[N,K]^T (bf16, row-major)
// Block: 8 waves -> 256(M) x 64(N); each wave owns TWO 16-row M-tiles so each
// LDS B fragment feeds two WMMAs (8 WMMAs / k-step / wave). The 64x32 B tile
// is shared by all waves: staged once per k-step into double-buffered LDS via
// async DMA (1 x B128 per thread), overlapping WMMA of chunk k with DMA of
// chunk k+1. B fragments are preloaded so ds_loads clause ahead of the WMMAs.
// ---------------------------------------------------------------------------
__global__ __launch_bounds__(256) void gemm_bf16_wmma(
    const bf16_t* __restrict__ A, const bf16_t* __restrict__ W,
    float* __restrict__ C, int M, int N, int K) {
  __shared__ __align__(16) bf16_t bbuf[2][64 * 32];

  const int tid  = threadIdx.x;
  const int wave = tid >> 5;
  const int lane = tid & 31;
  const int half = lane >> 4;
  const int l    = lane & 15;
  const int m0 = blockIdx.x * 256 + wave * 32;
  const int n0 = blockIdx.y * 64;

  // staging: thread t copies 16B of row n = t>>2, 8-elem packet part = t&3
  const int sn = tid >> 2, sp = tid & 3;
  const bf16_t* wsrc = W + (size_t)(n0 + sn) * K + sp * 8;
  bf16_t* sdst0 = &bbuf[0][sn * 32 + sp * 8];
  bf16_t* sdst1 = &bbuf[1][sn * 32 + sp * 8];

  v8f acc[2][4];
#pragma unroll
  for (int r = 0; r < 2; ++r)
#pragma unroll
    for (int j = 0; j < 4; ++j) acc[r][j] = v8f_zero();

  const bf16_t* arow0 = A + (size_t)(m0 + l) * K;
  const bf16_t* arow1 = A + (size_t)(m0 + 16 + l) * K;

  async_copy16(sdst0, wsrc);           // prologue: stage k0 = 0
  for (int k0 = 0; k0 < K; k0 += 32) {
    const int cur = (k0 >> 5) & 1;
    if (k0 + 32 < K) {                 // prefetch next B tile into other buffer
      async_copy16(cur ? sdst0 : sdst1, wsrc + k0 + 32);
      wait_async_le1();                // own packet of chunk k0 has landed
    } else {
      wait_async_0();
    }
    __syncthreads();                   // all packets of chunk k0 in LDS

    const int ka = k0 + half * 8;
    v16bf a0 = join16(*(const v8bf*)(arow0 + ka),
                      *(const v8bf*)(arow0 + ka + 16));
    v16bf a1 = join16(*(const v8bf*)(arow1 + ka),
                      *(const v8bf*)(arow1 + ka + 16));

    v16bf bfrag[4];
#pragma unroll
    for (int j = 0; j < 4; ++j) {
      const bf16_t* bp = &bbuf[cur][(j * 16 + l) * 32 + half * 8];
      bfrag[j] = join16(*(const v8bf*)bp, *(const v8bf*)(bp + 16));
    }
#pragma unroll
    for (int j = 0; j < 4; ++j)
      acc[0][j] = __builtin_amdgcn_wmma_f32_16x16x32_bf16(
          false, a0, false, bfrag[j], (short)0, acc[0][j], false, false);
#pragma unroll
    for (int j = 0; j < 4; ++j)
      acc[1][j] = __builtin_amdgcn_wmma_f32_16x16x32_bf16(
          false, a1, false, bfrag[j], (short)0, acc[1][j], false, false);

    __syncthreads();                   // reads done before buffer is reused
  }

  // C/D layout: VGPR vr -> row + vr + half*8; lane l -> col n0 + j*16 + l.
#pragma unroll
  for (int r = 0; r < 2; ++r) {
#pragma unroll
    for (int j = 0; j < 4; ++j) {
#pragma unroll
      for (int vr = 0; vr < 8; ++vr) {
        C[(size_t)(m0 + r * 16 + vr + half * 8) * N + n0 + j * 16 + l] =
            acc[r][j][vr];
      }
    }
  }
}

// ---------------------------------------------------------------------------
// RoPE + cast for Q: qf32 [B*T, H*D] -> qb [B, H, T, D] bf16
// ---------------------------------------------------------------------------
__global__ void rope_cast_q(const float* __restrict__ qin,
                            const float* __restrict__ cosb,
                            const float* __restrict__ sinb,
                            bf16_t* __restrict__ qout) {
  int idx  = blockIdx.x * blockDim.x + threadIdx.x;
  int pair = idx & 63;
  int r    = idx >> 6;
  int t    = r & (SEQ - 1);
  r >>= 11;
  int h = r & (N_HEAD - 1);
  int b = r >> 4;
  if (b >= BATCH) return;

  const float* src = qin + (size_t)(b * SEQ + t) * N_EMBD + h * HEAD_DIM + 2 * pair;
  float e = src[0], o = src[1];
  float c = cosb[t * 64 + pair], s = sinb[t * 64 + pair];
  bf16_t* dst = qout + ((size_t)(b * N_HEAD + h) * SEQ + t) * HEAD_DIM + 2 * pair;
  dst[0] = (bf16_t)(e * c - o * s);
  dst[1] = (bf16_t)(e * s + o * c);
}

// RoPE + cast for K: kf32 [B*T, KV*D] -> kb [B, KV, T, D] bf16
__global__ void rope_cast_k(const float* __restrict__ kin,
                            const float* __restrict__ cosb,
                            const float* __restrict__ sinb,
                            bf16_t* __restrict__ kout) {
  int idx  = blockIdx.x * blockDim.x + threadIdx.x;
  int pair = idx & 63;
  int r    = idx >> 6;
  int t    = r & (SEQ - 1);
  r >>= 11;
  int h = r & (N_KV - 1);
  int b = r >> 2;
  if (b >= BATCH) return;

  const float* src = kin + (size_t)(b * SEQ + t) * KV_C + h * HEAD_DIM + 2 * pair;
  float e = src[0], o = src[1];
  float c = cosb[t * 64 + pair], s = sinb[t * 64 + pair];
  bf16_t* dst = kout + ((size_t)(b * N_KV + h) * SEQ + t) * HEAD_DIM + 2 * pair;
  dst[0] = (bf16_t)(e * c - o * s);
  dst[1] = (bf16_t)(e * s + o * c);
}

// V: vf32 [B*T, KV*D] -> vt [B, KV, D, T] bf16 (PV B-operand reads 2 runs of 8
// contiguous keys per lane)
__global__ void v_cast_transpose(const float* __restrict__ vin,
                                 bf16_t* __restrict__ vt) {
  int idx = blockIdx.x * blockDim.x + threadIdx.x;
  int d = idx & (HEAD_DIM - 1);
  int r = idx >> 7;
  int t = r & (SEQ - 1);
  r >>= 11;
  int h = r & (N_KV - 1);
  int b = r >> 2;
  if (b >= BATCH) return;

  float val = vin[(size_t)(b * SEQ + t) * KV_C + h * HEAD_DIM + d];
  vt[((size_t)(b * N_KV + h) * HEAD_DIM + d) * SEQ + t] = (bf16_t)val;
}

// ---------------------------------------------------------------------------
// Flash attention, causal, GQA. Block = one (b, head) x 128 q rows
// (8 waves x 16-row q-tiles), so the 8 waves SHARE each 32-key K/V chunk.
// K (32x128) and V^T (128x32) chunks are async-DMA'd into double-buffered LDS
// (4 x B128 per thread per chunk -> s_wait_asynccnt 0x4 keeps one chunk in
// flight). Block-uniform key loop; fully-masked waves skip compute but keep
// the barriers. Per active chunk per wave: 8 WMMAs for S, online softmax via
// half-wave shuffle butterflies, P restaged through LDS (s_wait_dscnt) into
// A-operand layout, 8 WMMAs for O += P*V. Operand fragments are preloaded in
// groups so ds_loads clause ahead of the WMMA bursts.
// ---------------------------------------------------------------------------
__global__ __launch_bounds__(256) void flash_attn_wmma(
    const bf16_t* __restrict__ Qb, const bf16_t* __restrict__ Kb,
    const bf16_t* __restrict__ Vtb, bf16_t* __restrict__ Ob) {
  __shared__ __align__(16) bf16_t kbuf[2][32 * 128];   // [key_local][d]
  __shared__ __align__(16) bf16_t vbuf[2][128 * 32];   // [d][key_local]
  __shared__ __align__(16) bf16_t pbuf[8][16 * 32];

  const int tid  = threadIdx.x;
  const int wave = tid >> 5;
  const int lane = tid & 31;
  const int half = lane >> 4;
  const int l    = lane & 15;

  int blk  = blockIdx.x;           // B * H * (SEQ/128) = 1024 blocks
  int st   = blk & 15;             // 128-row super-tile within the sequence
  int bh   = blk >> 4;
  int head = bh & (N_HEAD - 1);
  int b    = bh >> 4;

  const int kvh = head / NREP;
  const int qrow0 = st * 128 + wave * 16;      // this wave's 16 q rows
  const bf16_t* Q  = Qb  + ((size_t)(b * N_HEAD + head) * SEQ + qrow0) * HEAD_DIM;
  const bf16_t* Kp = Kb  + (size_t)(b * N_KV + kvh) * SEQ * HEAD_DIM;
  const bf16_t* Vt = Vtb + (size_t)(b * N_KV + kvh) * HEAD_DIM * SEQ;

  // Q tile as 4 A-operands (d-chunks of 32), held for the whole loop.
  v16bf qa[4];
#pragma unroll
  for (int dc = 0; dc < 4; ++dc) {
    const bf16_t* qp = Q + (size_t)l * HEAD_DIM + dc * 32 + half * 8;
    qa[dc] = join16(*(const v8bf*)qp, *(const v8bf*)(qp + 16));
  }

  // staging assignments (per thread, per chunk): 2 packets of K + 2 of V
  const int koff = tid * 16;                       // K chunk is 8KB contiguous
  const int vd = tid >> 1, vp = tid & 1;           // V: row d, 32B half
  const bf16_t* vsrc = Vt + (size_t)vd * SEQ + vp * 16;

  v8f o[8];
#pragma unroll
  for (int j = 0; j < 8; ++j) o[j] = v8f_zero();
  float mrow[8], lrow[8];
#pragma unroll
  for (int vr = 0; vr < 8; ++vr) { mrow[vr] = -1e30f; lrow[vr] = 0.0f; }

  const float scale = 0.08838834764831845f;        // 1/sqrt(128)
  const int nkc = st * 4 + 4;                      // block-uniform trip count
  const int qlast = qrow0 + 15;

  // prologue: stage chunk 0 into buffer 0
  {
    const bf16_t* ks = Kp + koff;
    async_copy16(&kbuf[0][koff], ks);
    async_copy16(&kbuf[0][koff + 8], ks + 8);
    async_copy16(&vbuf[0][vd * 32 + vp * 16], vsrc);
    async_copy16(&vbuf[0][vd * 32 + vp * 16 + 8], vsrc + 8);
  }

  for (int kc = 0; kc < nkc; ++kc) {
    const int cur = kc & 1;
    const int kbase = kc * 32;
    if (kc + 1 < nkc) {                            // prefetch next chunk
      const int nb = cur ^ 1;
      const bf16_t* ks = Kp + (size_t)(kbase + 32) * HEAD_DIM + koff;
      async_copy16(&kbuf[nb][koff], ks);
      async_copy16(&kbuf[nb][koff + 8], ks + 8);
      const bf16_t* vs = vsrc + kbase + 32;
      async_copy16(&vbuf[nb][vd * 32 + vp * 16], vs);
      async_copy16(&vbuf[nb][vd * 32 + vp * 16 + 8], vs + 8);
      wait_async_le4();                            // chunk kc has landed
    } else {
      wait_async_0();
    }
    __syncthreads();                               // whole tile visible

    if (kbase <= qlast) {                          // causal: rows need chunk
      // ---- S = Q * K^T : two 16x16 tiles over this 32-key chunk ----
      v8f s[2];
      s[0] = v8f_zero(); s[1] = v8f_zero();
#pragma unroll
      for (int t2 = 0; t2 < 2; ++t2) {
        const bf16_t* kr = &kbuf[cur][(t2 * 16 + l) * HEAD_DIM + half * 8];
        v16bf bk[4];
#pragma unroll
        for (int dc = 0; dc < 4; ++dc)
          bk[dc] = join16(*(const v8bf*)(kr + dc * 32),
                          *(const v8bf*)(kr + dc * 32 + 16));
#pragma unroll
        for (int dc = 0; dc < 4; ++dc)
          s[t2] = __builtin_amdgcn_wmma_f32_16x16x32_bf16(
              false, qa[dc], false, bk[dc], (short)0, s[t2], false, false);
      }

      // ---- causal mask + online softmax (rows live across half-wave) ----
      float pr[2][8];
#pragma unroll
      for (int vr = 0; vr < 8; ++vr) {
        const int qrow = qrow0 + vr + half * 8;
#pragma unroll
        for (int t2 = 0; t2 < 2; ++t2) {
          const int key = kbase + t2 * 16 + l;
          float v = s[t2][vr] * scale;
          pr[t2][vr] = (key <= qrow) ? v : -1e30f;
        }
        float mx = fmaxf(pr[0][vr], pr[1][vr]);
#pragma unroll
        for (int m = 8; m >= 1; m >>= 1)
          mx = fmaxf(mx, __shfl_xor(mx, m, 32));

        float newm = fmaxf(mrow[vr], mx);
        float corr = __expf(mrow[vr] - newm);
        mrow[vr] = newm;
        float p0 = __expf(pr[0][vr] - newm);
        float p1 = __expf(pr[1][vr] - newm);
        pr[0][vr] = p0; pr[1][vr] = p1;
        float rs = p0 + p1;
#pragma unroll
        for (int m = 8; m >= 1; m >>= 1)
          rs += __shfl_xor(rs, m, 32);
        lrow[vr] = lrow[vr] * corr + rs;
#pragma unroll
        for (int j = 0; j < 8; ++j) o[j][vr] *= corr;
      }

      // ---- restage P through LDS: C-layout -> row-major -> A-operand ----
#pragma unroll
      for (int vr = 0; vr < 8; ++vr) {
        pbuf[wave][(vr + half * 8) * 32 + l]      = (bf16_t)pr[0][vr];
        pbuf[wave][(vr + half * 8) * 32 + 16 + l] = (bf16_t)pr[1][vr];
      }
      asm volatile("s_wait_dscnt 0x0" ::: "memory");  // per-wave store->load
      const bf16_t* pw = &pbuf[wave][(size_t)l * 32 + half * 8];
      v16bf pa = join16(*(const v8bf*)pw, *(const v8bf*)(pw + 16));

      // ---- O += P * V over 8 d-chunks of 16, fragments in groups of 4 ----
#pragma unroll
      for (int jj = 0; jj < 8; jj += 4) {
        v16bf bv[4];
#pragma unroll
        for (int j2 = 0; j2 < 4; ++j2) {
          const bf16_t* vpp = &vbuf[cur][((jj + j2) * 16 + l) * 32 + half * 8];
          bv[j2] = join16(*(const v8bf*)vpp, *(const v8bf*)(vpp + 16));
        }
#pragma unroll
        for (int j2 = 0; j2 < 4; ++j2)
          o[jj + j2] = __builtin_amdgcn_wmma_f32_16x16x32_bf16(
              false, pa, false, bv[j2], (short)0, o[jj + j2], false, false);
      }
    }
    __syncthreads();                               // reads done before reuse
  }

  // ---- epilogue: O /= l, store bf16 to attn [B*T, H*D] ----
#pragma unroll
  for (int vr = 0; vr < 8; ++vr) {
    float inv = 1.0f / lrow[vr];
    size_t row = (size_t)(b * SEQ + qrow0 + vr + half * 8);
    bf16_t* dst = Ob + row * N_EMBD + head * HEAD_DIM + l;
#pragma unroll
    for (int j = 0; j < 8; ++j)
      dst[j * 16] = (bf16_t)(o[j][vr] * inv);
  }
}

// ---------------------------------------------------------------------------
// Host-side orchestration
// ---------------------------------------------------------------------------
extern "C" void kernel_launch(void* const* d_in, const int* in_sizes, int n_in,
                              void* d_out, int out_size, void* d_ws, size_t ws_size,
                              hipStream_t stream) {
  (void)in_sizes; (void)n_in; (void)out_size; (void)ws_size;
  const float* x  = (const float*)d_in[0];
  const float* wq = (const float*)d_in[1];
  const float* wk = (const float*)d_in[2];
  const float* wv = (const float*)d_in[3];
  const float* wo = (const float*)d_in[4];
  const float* rc = (const float*)d_in[5];
  const float* rs = (const float*)d_in[6];
  float* out = (float*)d_out;

  const size_t M = (size_t)BATCH * SEQ;   // 8192 rows
  const size_t C = N_EMBD;                // 2048
  const size_t KV = KV_C;                 // 512

  char* p = (char*)d_ws;
  auto alloc = [&](size_t bytes) -> char* {
    char* r = p;
    p += (bytes + 255) & ~(size_t)255;
    return r;
  };
  bf16_t* x_bf  = (bf16_t*)alloc(M * C * 2);
  bf16_t* wq_bf = (bf16_t*)alloc(C * C * 2);
  bf16_t* wk_bf = (bf16_t*)alloc(KV * C * 2);
  bf16_t* wv_bf = (bf16_t*)alloc(KV * C * 2);
  bf16_t* wo_bf = (bf16_t*)alloc(C * C * 2);
  float*  qf    = (float*)alloc(M * C * 4);
  float*  kf    = (float*)alloc(M * KV * 4);
  float*  vf    = (float*)alloc(M * KV * 4);
  bf16_t* qb    = (bf16_t*)alloc(M * C * 2);
  bf16_t* kb    = (bf16_t*)alloc(M * KV * 2);
  bf16_t* vtb   = (bf16_t*)alloc(M * KV * 2);
  bf16_t* ab    = (bf16_t*)alloc(M * C * 2);

  const int TB = 256;
  auto blocks = [](size_t n, int tb) { return (unsigned)((n + tb - 1) / tb); };

  // 1) cast everything to bf16 once
  cast_f32_to_bf16<<<blocks(M * C, TB), TB, 0, stream>>>(x,  x_bf,  (int)(M * C));
  cast_f32_to_bf16<<<blocks(C * C, TB), TB, 0, stream>>>(wq, wq_bf, (int)(C * C));
  cast_f32_to_bf16<<<blocks(KV * C, TB), TB, 0, stream>>>(wk, wk_bf, (int)(KV * C));
  cast_f32_to_bf16<<<blocks(KV * C, TB), TB, 0, stream>>>(wv, wv_bf, (int)(KV * C));
  cast_f32_to_bf16<<<blocks(C * C, TB), TB, 0, stream>>>(wo, wo_bf, (int)(C * C));

  // 2) QKV projections on the WMMA units (async-LDS-staged B tiles)
  gemm_bf16_wmma<<<dim3(M / 256, C / 64), TB, 0, stream>>>(x_bf, wq_bf, qf,
                                                           (int)M, (int)C, (int)C);
  gemm_bf16_wmma<<<dim3(M / 256, KV / 64), TB, 0, stream>>>(x_bf, wk_bf, kf,
                                                            (int)M, (int)KV, (int)C);
  gemm_bf16_wmma<<<dim3(M / 256, KV / 64), TB, 0, stream>>>(x_bf, wv_bf, vf,
                                                            (int)M, (int)KV, (int)C);

  // 3) RoPE + layout transforms (q/k rotated, v transposed to [B,KV,D,T])
  rope_cast_q<<<blocks((size_t)BATCH * N_HEAD * SEQ * 64, TB), TB, 0, stream>>>(qf, rc, rs, qb);
  rope_cast_k<<<blocks((size_t)BATCH * N_KV * SEQ * 64, TB), TB, 0, stream>>>(kf, rc, rs, kb);
  v_cast_transpose<<<blocks((size_t)BATCH * N_KV * SEQ * HEAD_DIM, TB), TB, 0, stream>>>(vf, vtb);

  // 4) causal flash attention: block = (b, head, 128-row super-tile)
  flash_attn_wmma<<<(unsigned)(BATCH * N_HEAD * (SEQ / 128)), TB, 0, stream>>>(qb, kb, vtb, ab);

  // 5) output projection -> d_out (fp32)
  gemm_bf16_wmma<<<dim3(M / 256, C / 64), TB, 0, stream>>>(ab, wo_bf, out,
                                                           (int)M, (int)C, (int)C);
}
